// VectorQuantizer_20942260535677
// MI455X (gfx1250) — matrix operand order repaired and checked by reading, not compiled, not run on hardware
//
#include <hip/hip_runtime.h>
#include <hip/hip_fp16.h>

typedef __attribute__((ext_vector_type(16))) _Float16 v16h;
typedef __attribute__((ext_vector_type(8)))  float    v8f;
typedef int v4i_ __attribute__((vector_size(16)));   // matches builtin param type

#define D_TOTAL 131072
#define L_DIM   64
#define K_CB    2048
#define ROW_PAD 72            // halves per padded LDS row (144 B, 16B-aligned, breaks bank conflicts)
#define ROWS_PER_BLOCK 256    // 8 waves * 32 rows
#define EPSN 1e-8f

#if defined(__gfx1250__) && __has_builtin(__builtin_amdgcn_global_load_async_to_lds_b128) && __has_builtin(__builtin_amdgcn_s_wait_asynccnt)
#define USE_ASYNC_CB_STAGE 1
#else
#define USE_ASYNC_CB_STAGE 0
#endif

union V16U { v16h h; float4 f[2]; };

// ---------------- Kernel 1: normalize codebook -> f16 workspace ----------------
__global__ void vq_normalize_cb(const float* __restrict__ cb, _Float16* __restrict__ cb16) {
    int row = blockIdx.x * blockDim.x + threadIdx.x;
    if (row >= K_CB) return;
    const float* src = cb + (size_t)row * L_DIM;
    float v[L_DIM];
    float ss = 0.f;
#pragma unroll
    for (int i = 0; i < L_DIM; ++i) { v[i] = src[i]; ss += v[i] * v[i]; }
    float scale = 1.0f / (sqrtf(ss) + EPSN);
    _Float16* dst = cb16 + (size_t)row * L_DIM;
#pragma unroll
    for (int i = 0; i < L_DIM; ++i) dst[i] = (_Float16)(v[i] * scale);
}

// ---------------- Kernel 2: normalize x, WMMA similarity, argmax, gather -------
__launch_bounds__(256, 1)
__global__ void vq_main(const float* __restrict__ x,
                        const float* __restrict__ cb,
                        const _Float16* __restrict__ cb16,
                        float* __restrict__ zq,
                        float* __restrict__ z,
                        float* __restrict__ xn,
                        float* __restrict__ idx_out) {
    extern __shared__ char smem[];
    _Float16* lds_cb = (_Float16*)smem;                                // [K_CB][ROW_PAD]
    _Float16* lds_x  = (_Float16*)(smem + (size_t)K_CB * ROW_PAD * 2); // per-wave [16][ROW_PAD]

    const int tid  = threadIdx.x;
    const int wave = tid >> 5;
    const int lane = tid & 31;
    const int sl   = lane & 15;   // sub-lane within 16-group
    const int hi   = lane >> 4;   // which half of the wave

    // ---- stage whole f16 codebook into LDS (padded rows) ----
#if USE_ASYNC_CB_STAGE
    for (int r = tid; r < K_CB; r += 256) {
        const float4* s4 = (const float4*)(cb16 + (size_t)r * L_DIM);
        float4* d4 = (float4*)(lds_cb + (size_t)r * ROW_PAD);
#pragma unroll
        for (int i = 0; i < 8; ++i) {
            __builtin_amdgcn_global_load_async_to_lds_b128(
                (__attribute__((address_space(1))) v4i_*)(s4 + i),
                (__attribute__((address_space(3))) v4i_*)(d4 + i),
                /*offset=*/0, /*cpol=*/0);
        }
    }
    __builtin_amdgcn_s_wait_asynccnt(0);
#else
    for (int r = tid; r < K_CB; r += 256) {
        const float4* s4 = (const float4*)(cb16 + (size_t)r * L_DIM);
        float4* d4 = (float4*)(lds_cb + (size_t)r * ROW_PAD);
#pragma unroll
        for (int i = 0; i < 8; ++i) d4[i] = s4[i];   // 64 halves = 8 x 16B
    }
#endif

    // ---- per-wave: normalize 32 x-rows (two 16-row tiles), build 2 A-fragment
    // sets. The 16-row LDS staging buffer is reused for both tiles (same-wave
    // LDS ops are in-order, no barrier needed within the wave).
    const int row_base = blockIdx.x * ROWS_PER_BLOCK + wave * 32;
    _Float16* xstage = lds_x + (size_t)wave * 16 * ROW_PAD;
    const int o = hi * 8;
    V16U a[4];   // a[0],a[1] = row tile 0 (k 0..31 / 32..63); a[2],a[3] = row tile 1

#pragma unroll
    for (int t = 0; t < 2; ++t) {
        const int r  = t * 16 + sl;   // row within the wave's 32-row strip
        const int c0 = hi * 32;       // this lane handles 32 of the 64 columns
        const float* src = x + (size_t)(row_base + r) * L_DIM + c0;
        float v[32];
        float ss = 0.f;
#pragma unroll
        for (int i = 0; i < 32; ++i) { v[i] = src[i]; ss += v[i] * v[i]; }
        ss += __shfl_xor(ss, 16, 32);                 // combine the two half-rows
        float scale = 1.0f / (sqrtf(ss) + EPSN);
        float* xd = xn + (size_t)(row_base + r) * L_DIM + c0;
        _Float16* ld = xstage + (size_t)sl * ROW_PAD + c0;
#pragma unroll
        for (int i = 0; i < 32; ++i) {
            float nv = v[i] * scale;
            xd[i] = nv;
            ld[i] = (_Float16)nv;
        }
        // A fragment layout (16-bit A 16x32):
        // lane<16: halves 0..7 = K 0..7, 8..15 = K 16..23 ; lane>=16: K 8..15 / 24..31
        const _Float16* xrow = xstage + (size_t)sl * ROW_PAD;
        a[2 * t + 0].f[0] = *(const float4*)(xrow + o);
        a[2 * t + 0].f[1] = *(const float4*)(xrow + 16 + o);
        a[2 * t + 1].f[0] = *(const float4*)(xrow + 32 + o);
        a[2 * t + 1].f[1] = *(const float4*)(xrow + 48 + o);
    }
    __syncthreads();   // codebook LDS visible to all waves

    // ---- sweep K in 16-column tiles: one B-fragment pair feeds 4 WMMAs
    // (two independent accumulator chains -> scheduler hides WMMA->VALU NOPs) ----
    float bestv[2][8];
    int   besti[2][8];
#pragma unroll
    for (int t = 0; t < 2; ++t)
#pragma unroll
        for (int j = 0; j < 8; ++j) { bestv[t][j] = -3.0e38f; besti[t][j] = 0; }

    for (int kt = 0; kt < K_CB / 16; ++kt) {
        const int col = kt * 16 + sl;                // this lane's output column = codebook row
        const _Float16* crow = lds_cb + (size_t)col * ROW_PAD + hi * 8;
        V16U b0, b1;
        b0.f[0] = ((const float4*)crow)[0];          // K = ko..ko+7, 16+ko..16+ko+7   (ko = hi*8)
        b0.f[1] = ((const float4*)crow)[2];
        b1.f[0] = ((const float4*)(crow + 32))[0];   // K = 32+ko..32+ko+7, 48+ko..48+ko+7
        b1.f[1] = ((const float4*)(crow + 32))[2];

        v8f c0 = {}, c1 = {};
        c0 = __builtin_amdgcn_wmma_f32_16x16x32_f16(false, a[0].h, false, b0.h,
                                                    (short)0, c0, false, false);
        c1 = __builtin_amdgcn_wmma_f32_16x16x32_f16(false, a[2].h, false, b0.h,
                                                    (short)0, c1, false, false);
        c0 = __builtin_amdgcn_wmma_f32_16x16x32_f16(false, a[1].h, false, b1.h,
                                                    (short)0, c0, false, false);
        c1 = __builtin_amdgcn_wmma_f32_16x16x32_f16(false, a[3].h, false, b1.h,
                                                    (short)0, c1, false, false);
#pragma unroll
        for (int j = 0; j < 8; ++j) {
            bool b0t = c0[j] > bestv[0][j];          // strict > keeps earliest index
            bestv[0][j] = b0t ? c0[j] : bestv[0][j];
            besti[0][j] = b0t ? col : besti[0][j];
            bool b1t = c1[j] > bestv[1][j];
            bestv[1][j] = b1t ? c1[j] : bestv[1][j];
            besti[1][j] = b1t ? col : besti[1][j];
        }
    }

    // ---- cross-lane argmax within each 16-lane group (C layout: slot j = row) ----
#pragma unroll
    for (int off = 8; off; off >>= 1) {
#pragma unroll
        for (int t = 0; t < 2; ++t)
#pragma unroll
            for (int j = 0; j < 8; ++j) {
                float ov = __shfl_xor(bestv[t][j], off, 16);
                int   oi = __shfl_xor(besti[t][j], off, 16);
                bool take = (ov > bestv[t][j]) || (ov == bestv[t][j] && oi < besti[t][j]);
                bestv[t][j] = take ? ov : bestv[t][j];
                besti[t][j] = take ? oi : besti[t][j];
            }
    }

    // ---- emit indices + gather unnormalized codebook rows into z_q and z ----
    // lanes 0..15 own rows 0..7 (slots 0..7); lanes 16..31 own rows 8..15.
#pragma unroll
    for (int t = 0; t < 2; ++t) {
#pragma unroll
        for (int j = 0; j < 8; ++j) {
            const int grow = row_base + t * 16 + hi * 8 + j;
            const int idx  = besti[t][j];            // uniform across the 16-lane group
            if (sl == 0) idx_out[grow] = (float)idx;
            float4 v4 = ((const float4*)(cb + (size_t)idx * L_DIM))[sl];
            ((float4*)(zq + (size_t)grow * L_DIM))[sl] = v4;
            ((float4*)(z  + (size_t)grow * L_DIM))[sl] = v4;
        }
    }
}

extern "C" void kernel_launch(void* const* d_in, const int* in_sizes, int n_in,
                              void* d_out, int out_size, void* d_ws, size_t ws_size,
                              hipStream_t stream) {
    const float* x  = (const float*)d_in[0];   // [131072, 64]
    const float* cb = (const float*)d_in[1];   // [2048, 64]
    // d_in[2] = training flag (unused; reference path is inference)

    float* out  = (float*)d_out;
    float* zq   = out;                                     // [D, L]
    float* z    = out + (size_t)D_TOTAL * L_DIM;           // [D, L]
    float* xn   = out + (size_t)2 * D_TOTAL * L_DIM;       // [D, L]
    float* idxf = out + (size_t)3 * D_TOTAL * L_DIM;       // [D]

    _Float16* cb16 = (_Float16*)d_ws;                      // 2048*64*2 = 256 KB

    vq_normalize_cb<<<K_CB / 256, 256, 0, stream>>>(cb, cb16);

    const size_t lds_bytes = (size_t)K_CB * ROW_PAD * 2    // 288 KB codebook
                           + (size_t)8 * 16 * ROW_PAD * 2; // 18 KB x staging  (306 KB < 320 KB WGP LDS)
    vq_main<<<D_TOTAL / ROWS_PER_BLOCK, 256, lds_bytes, stream>>>(
        x, cb, cb16, zq, z, xn, idxf);
}
